// BidirectionalAttention_39977555591434
// MI455X (gfx1250) — matrix operand order repaired and checked by reading, not compiled, not run on hardware
//
#include <hip/hip_runtime.h>
#include <cstdint>
#include <cstddef>

// ---------------------------------------------------------------------------
// Types for CDNA5 WMMA
// ---------------------------------------------------------------------------
typedef __bf16 bf16;
typedef __attribute__((ext_vector_type(16))) __bf16   v16bf;
typedef __attribute__((ext_vector_type(8)))  float    v8f;
typedef __attribute__((ext_vector_type(4)))  uint32_t uvec4;   // 16-byte chunk

#define LDSPITCH 40   // halves per LDS tile row: 32 data + 8 pad (80B rows)

// Branch-free tanh: sign(x) * (1 - e^(-2|x|)) / (1 + e^(-2|x|)).
// e in (0,1] so no overflow / NaN paths; avoids ocml's divergent libm tanh.
__device__ __forceinline__ float fast_tanh(float x) {
  float a = __builtin_fabsf(x);
  float e = __expf(-2.0f * a);
  float r = (1.0f - e) / (1.0f + e);
  return __builtin_copysignf(r, x);
}

// One 16-byte async global->LDS copy (CDNA5, ASYNCcnt-tracked).
__device__ __forceinline__ void async_cp16(uint32_t lds_byte_off, const bf16* g) {
  asm volatile("global_load_async_to_lds_b128 %0, %1, off"
               :: "v"(lds_byte_off), "v"(g) : "memory");
}
__device__ __forceinline__ void wait_async0() {
  asm volatile("s_wait_asynccnt 0x0" ::: "memory");
}

// ---------------------------------------------------------------------------
// Generic bf16 WMMA GEMM:  C[M,N] = act( A[M,K] @ Bt[N,K]^T )
//   - A row-major (lda), Bt stored as N rows of K (ldb)  -> symmetric frags
//   - block tile 128x128, BK=32, 256 threads = 8 waves, wave tile 32x64
//   - double-buffered LDS fed by GLOBAL_LOAD_ASYNC_TO_LDS_B128
//   - all dims assumed multiples of the tile sizes (true for this problem)
// ---------------------------------------------------------------------------
template <typename OutT, bool TANH>
__global__ __launch_bounds__(256) void gemm_bf16_wmma(
    const bf16* __restrict__ A, long long strideA, int lda,
    const bf16* __restrict__ B, long long strideB, int ldb,
    OutT* __restrict__ C, long long strideC, int ldc,
    int K)
{
  __shared__ bf16 ldsA[2][128 * LDSPITCH];
  __shared__ bf16 ldsB[2][128 * LDSPITCH];

  const int z = blockIdx.z;
  A += (long long)z * strideA;
  B += (long long)z * strideB;
  C += (long long)z * strideC;

  const int tid   = threadIdx.x;
  const int lane  = tid & 31;
  const int w     = tid >> 5;
  const int waveM = w & 3;            // 0..3  (M direction, 32 rows each)
  const int waveN = w >> 2;           // 0..1  (N direction, 64 cols each)
  const int m0 = blockIdx.y * 128;
  const int n0 = blockIdx.x * 128;

  // staging map: each thread moves 32 bytes (16 halves) of A and of B
  const int srow = tid >> 1;          // 0..127
  const int scol = (tid & 1) * 16;    // 0 or 16 (halves)

  // fragment maps (per ISA 7.12.2 tables, wave32)
  const int fr  = lane & 15;          // row-in-tile for A, col-in-tile for B
  const int ka  = (lane >> 4) * 8;    // A: chunks at ka and ka+16
  const int kb2 = (lane >> 4) * 16;   // unused; B also reads chunks at ka and ka+16

  // per-thread LDS byte offsets for staging (flat addr low 32 bits = LDS off)
  const uint32_t ldsAoff[2] = {
      (uint32_t)(uintptr_t)&ldsA[0][srow * LDSPITCH + scol],
      (uint32_t)(uintptr_t)&ldsA[1][srow * LDSPITCH + scol] };
  const uint32_t ldsBoff[2] = {
      (uint32_t)(uintptr_t)&ldsB[0][srow * LDSPITCH + scol],
      (uint32_t)(uintptr_t)&ldsB[1][srow * LDSPITCH + scol] };
  const bf16* gArow = A + (size_t)(m0 + srow) * lda + scol;
  const bf16* gBrow = B + (size_t)(n0 + srow) * ldb + scol;

  v8f acc[2][4] = {};                 // 64 VGPRs of f32 accumulators

  // prologue: kick off K-tile 0 into buffer 0
  async_cp16(ldsAoff[0],      gArow);
  async_cp16(ldsAoff[0] + 16, gArow + 8);
  async_cp16(ldsBoff[0],      gBrow);
  async_cp16(ldsBoff[0] + 16, gBrow + 8);

  for (int k0 = 0; k0 < K; k0 += 32) {
    const int buf = (k0 >> 5) & 1;

    // my async batch for `buf` has landed; barrier => everyone's has, and
    // every wave is done reading buf^1 (its compute finished last iteration)
    wait_async0();
    __syncthreads();

    // prefetch next K-tile into the other buffer, overlapped with the WMMAs
    if (k0 + 32 < K) {
      const bf16* ga = gArow + (k0 + 32);
      const bf16* gb = gBrow + (k0 + 32);
      async_cp16(ldsAoff[buf ^ 1],      ga);
      async_cp16(ldsAoff[buf ^ 1] + 16, ga + 8);
      async_cp16(ldsBoff[buf ^ 1],      gb);
      async_cp16(ldsBoff[buf ^ 1] + 16, gb + 8);
    }

    // ---- load fragments from LDS ----
    v16bf afrag[2], bfrag[4];
#pragma unroll
    for (int mt = 0; mt < 2; ++mt) {
      const bf16* p = &ldsA[buf][(waveM * 32 + mt * 16 + fr) * LDSPITCH + ka];
      union { v16bf v; uvec4 u[2]; } t;
      t.u[0] = *(const uvec4*)(p);
      t.u[1] = *(const uvec4*)(p + 16);
      afrag[mt] = t.v;
    }
#pragma unroll
    for (int nt = 0; nt < 4; ++nt) {
      const bf16* p = &ldsB[buf][(waveN * 64 + nt * 16 + fr) * LDSPITCH + ka];
      union { v16bf v; uvec4 u[2]; } t;
      t.u[0] = *(const uvec4*)(p);
      t.u[1] = *(const uvec4*)(p + 16);
      bfrag[nt] = t.v;
    }

    // ---- 8 WMMA ops per wave per K-step ----
#pragma unroll
    for (int mt = 0; mt < 2; ++mt)
#pragma unroll
      for (int nt = 0; nt < 4; ++nt)
        acc[mt][nt] = __builtin_amdgcn_wmma_f32_16x16x32_bf16(
            false, afrag[mt], false, bfrag[nt],
            (short)0, acc[mt][nt], false, false);
  }

  // ---- epilogue: C/D layout = {row r + (lane>=16)*8, col lane&15} ----
  const int cn   = lane & 15;
  const int rAdd = (lane >> 4) * 8;
#pragma unroll
  for (int mt = 0; mt < 2; ++mt) {
#pragma unroll
    for (int nt = 0; nt < 4; ++nt) {
      const int col     = n0 + waveN * 64 + nt * 16 + cn;
      const int rowBase = m0 + waveM * 32 + mt * 16 + rAdd;
      v8f v = acc[mt][nt];
#pragma unroll
      for (int r = 0; r < 8; ++r) {
        float x = v[r];
        if (TANH) x = fast_tanh(x);
        C[(size_t)(rowBase + r) * ldc + col] = (OutT)x;
      }
    }
  }
}

// ---------------------------------------------------------------------------
// f32 -> bf16 convert, fused with the concat copy into the output tensor
// ---------------------------------------------------------------------------
__global__ __launch_bounds__(256) void cvt_copy_kernel(
    const float* __restrict__ X, bf16* __restrict__ Xb,
    float* __restrict__ outCopy, int total, int d, int ldOut)
{
  int idx = blockIdx.x * 256 + threadIdx.x;
  if (idx >= total) return;
  float v = X[idx];
  Xb[idx] = (bf16)v;
  int r = idx / d;
  int c = idx - r * d;
  outCopy[(size_t)r * ldOut + c] = v;
}

__global__ __launch_bounds__(256) void cvt_kernel(
    const float* __restrict__ X, bf16* __restrict__ Xb, int total)
{
  int idx = blockIdx.x * 256 + threadIdx.x;
  if (idx < total) Xb[idx] = (bf16)X[idx];
}

// ---------------------------------------------------------------------------
// Per-batch transpose with conversion:  XT[n][d][s] = (bf16) X[n][s][d]
// ---------------------------------------------------------------------------
__global__ __launch_bounds__(256) void transpose_cvt_kernel(
    const float* __restrict__ X, bf16* __restrict__ XT, int S, int D)
{
  __shared__ float tile[32][33];
  const int n = blockIdx.z;
  const float* src = X + (size_t)n * S * D;
  bf16* dst = XT + (size_t)n * S * D;
  const int s0 = blockIdx.x * 32;
  const int d0 = blockIdx.y * 32;
#pragma unroll
  for (int i = threadIdx.y; i < 32; i += 8)
    tile[i][threadIdx.x] = src[(size_t)(s0 + i) * D + d0 + threadIdx.x];
  __syncthreads();
#pragma unroll
  for (int i = threadIdx.y; i < 32; i += 8)
    dst[(size_t)(d0 + i) * S + s0 + threadIdx.x] = (bf16)tile[threadIdx.x][i];
}

// ---------------------------------------------------------------------------
// Softmax over rows:  P[row,:] = softmax(S[row,:] * scale), bf16 out
// one 256-thread block per row, cols == 2048 (8 elems/thread)
// ---------------------------------------------------------------------------
__global__ __launch_bounds__(256) void softmax_rows_kernel(
    const float* __restrict__ S, bf16* __restrict__ P, int cols, float scale)
{
  __shared__ float red[256];
  const size_t row = blockIdx.x;
  const float* src = S + row * cols;
  bf16* dst = P + row * cols;
  const int tid = threadIdx.x;

  float x[8];
  float m = -3.0e38f;
#pragma unroll
  for (int i = 0; i < 8; ++i) {
    x[i] = src[tid + i * 256] * scale;
    m = fmaxf(m, x[i]);
  }
  red[tid] = m; __syncthreads();
  for (int s = 128; s > 0; s >>= 1) {
    if (tid < s) red[tid] = fmaxf(red[tid], red[tid + s]);
    __syncthreads();
  }
  m = red[0]; __syncthreads();

  float sum = 0.f;
#pragma unroll
  for (int i = 0; i < 8; ++i) { x[i] = __expf(x[i] - m); sum += x[i]; }
  red[tid] = sum; __syncthreads();
  for (int s = 128; s > 0; s >>= 1) {
    if (tid < s) red[tid] += red[tid + s];
    __syncthreads();
  }
  const float inv = 1.0f / red[0];
#pragma unroll
  for (int i = 0; i < 8; ++i) dst[tid + i * 256] = (bf16)(x[i] * inv);
}

// ---------------------------------------------------------------------------
// Softmax over columns, writing the TRANSPOSED result:
//   PT[n][t][s] = softmax_s(S[n][s][t] * scale)
// one 256-thread block per (n,t); rows == 2048 (8 elems/thread)
// ---------------------------------------------------------------------------
__global__ __launch_bounds__(256) void softmax_cols_kernel(
    const float* __restrict__ S, bf16* __restrict__ PT,
    int rows, int cols, float scale)
{
  __shared__ float red[256];
  const int n = blockIdx.x / cols;
  const int t = blockIdx.x - n * cols;
  const float* src = S + (size_t)n * rows * cols + t;
  bf16* dst = PT + ((size_t)n * cols + t) * rows;
  const int tid = threadIdx.x;

  float x[8];
  float m = -3.0e38f;
#pragma unroll
  for (int i = 0; i < 8; ++i) {
    x[i] = src[(size_t)(tid + i * 256) * cols] * scale;
    m = fmaxf(m, x[i]);
  }
  red[tid] = m; __syncthreads();
  for (int s = 128; s > 0; s >>= 1) {
    if (tid < s) red[tid] = fmaxf(red[tid], red[tid + s]);
    __syncthreads();
  }
  m = red[0]; __syncthreads();

  float sum = 0.f;
#pragma unroll
  for (int i = 0; i < 8; ++i) { x[i] = __expf(x[i] - m); sum += x[i]; }
  red[tid] = sum; __syncthreads();
  for (int s = 128; s > 0; s >>= 1) {
    if (tid < s) red[tid] += red[tid + s];
    __syncthreads();
  }
  const float inv = 1.0f / red[0];
#pragma unroll
  for (int i = 0; i < 8; ++i) dst[tid + i * 256] = (bf16)(x[i] * inv);
}

// ---------------------------------------------------------------------------
// Host-side orchestration
// ---------------------------------------------------------------------------
extern "C" void kernel_launch(void* const* d_in, const int* in_sizes, int n_in,
                              void* d_out, int out_size, void* d_ws, size_t ws_size,
                              hipStream_t stream)
{
  (void)in_sizes; (void)n_in; (void)out_size; (void)ws_size;
  const int  NB = 8, S = 2048, D = 1024;
  const float scale = 1.0f / 32.0f;     // 1/sqrt(1024)

  const float* lhs = (const float*)d_in[0];
  const float* rhs = (const float*)d_in[1];
  const float* Wl  = (const float*)d_in[2];
  const float* Wr  = (const float*)d_in[3];

  float* out_lhs = (float*)d_out;                       // (8,2048,2048)
  float* out_rhs = out_lhs + (size_t)NB * S * (2 * D);  // (8,2048,2048)

  // ---- workspace layout (bf16 unless noted) ----
  char* ws = (char*)d_ws;
  const size_t szX = (size_t)NB * S * D * sizeof(bf16);   // 33.5 MB
  const size_t szW = (size_t)D * D * sizeof(bf16);        // 2 MB
  bf16*  Xl  = (bf16*)(ws + 0 * szX);
  bf16*  Xr  = (bf16*)(ws + 1 * szX);
  bf16*  Wlb = (bf16*)(ws + 2 * szX);
  bf16*  Wrb = (bf16*)(ws + 2 * szX + szW);
  bf16*  XlT = (bf16*)(ws + 2 * szX + 2 * szW);
  bf16*  XrT = (bf16*)(ws + 3 * szX + 2 * szW);
  bf16*  L   = (bf16*)(ws + 4 * szX + 2 * szW);
  bf16*  R   = (bf16*)(ws + 5 * szX + 2 * szW);
  float* Sc  = (float*)(ws + 6 * szX + 2 * szW);          // 134 MB f32 scores
  // dead-region reuse (sequential launches on one stream):
  bf16*  attnL  = L;    // 67 MB over [L,R]   (L,R dead after scores GEMM)
  bf16*  attnRT = Xl;   // 67 MB over [Xl,Xr] (Xl,Xr dead after projections)

  const int totX = NB * S * D;       // 16,777,216
  const int totW = D * D;            // 1,048,576

  // 1) convert inputs to bf16 + fused concat copy of raw inputs into outputs
  cvt_copy_kernel<<<(totX + 255) / 256, 256, 0, stream>>>(lhs, Xl, out_lhs, totX, D, 2 * D);
  cvt_copy_kernel<<<(totX + 255) / 256, 256, 0, stream>>>(rhs, Xr, out_rhs, totX, D, 2 * D);
  cvt_kernel<<<(totW + 255) / 256, 256, 0, stream>>>(Wl, Wlb, totW);
  cvt_kernel<<<(totW + 255) / 256, 256, 0, stream>>>(Wr, Wrb, totW);

  // 2) per-batch transposed bf16 copies of lhs/rhs for the context GEMMs
  transpose_cvt_kernel<<<dim3(S / 32, D / 32, NB), dim3(32, 8), 0, stream>>>(lhs, XlT, S, D);
  transpose_cvt_kernel<<<dim3(S / 32, D / 32, NB), dim3(32, 8), 0, stream>>>(rhs, XrT, S, D);

  // 3) projections: L = tanh(X @ W^T)   (M=16384, N=1024, K=1024)
  {
    dim3 g(D / 128, (NB * S) / 128, 1);
    gemm_bf16_wmma<bf16, true><<<g, 256, 0, stream>>>(Xl, 0, D, Wlb, 0, D, L, 0, D, D);
    gemm_bf16_wmma<bf16, true><<<g, 256, 0, stream>>>(Xr, 0, D, Wrb, 0, D, R, 0, D, D);
  }

  // 4) scores[n] = L[n] @ R[n]^T   (per batch: M=N=2048, K=1024, f32 out)
  {
    dim3 g(S / 128, S / 128, NB);
    gemm_bf16_wmma<float, false><<<g, 256, 0, stream>>>(
        L, (long long)S * D, D, R, (long long)S * D, D,
        Sc, (long long)S * S, S, D);
  }

  // 5) softmaxes: row-wise -> attnL ; column-wise (transposed out) -> attnRT
  softmax_rows_kernel<<<NB * S, 256, 0, stream>>>(Sc, attnL, S, scale);
  softmax_cols_kernel<<<NB * S, 256, 0, stream>>>(Sc, attnRT, S, S, scale);

  // 6) contexts straight into the concat halves (ldc = 2048, col offset D)
  {
    dim3 g(D / 128, S / 128, NB);
    gemm_bf16_wmma<float, false><<<g, 256, 0, stream>>>(
        attnL, (long long)S * S, S, XrT, (long long)D * S, S,
        out_lhs + D, (long long)S * 2 * D, 2 * D, S);
    gemm_bf16_wmma<float, false><<<g, 256, 0, stream>>>(
        attnRT, (long long)S * S, S, XlT, (long long)D * S, S,
        out_rhs + D, (long long)S * 2 * D, 2 * D, S);
  }
}
